// GGNN_111669150309
// MI455X (gfx1250) — matrix operand order repaired and checked by reading, not compiled, not run on hardware
//
#include <hip/hip_runtime.h>
#include <hip/hip_bf16.h>

// ---------------------------------------------------------------------------
// GGNN on MI455X (gfx1250, wave32, WMMA + async-to-LDS)
//   per layer:  m = h @ W[l]            (bf16 WMMA, f32 acc, LDS-staged B)
//               agg = scatter_add(m[src]*ew, dst)   (f32 atomics, L2-resident)
//               h   = GRUCell(agg, h)   (6 fused bf16 WMMA chains per tile)
//   final:      out = relu(h) @ fc_w.T + fc_b
// K padded to 1024 (zero tail), nodes padded to 20096 = 157*128 so the
// 8-wave row blocking divides evenly (pad rows are row-independent garbage).
// Weight tiles are staged into double-buffered LDS with
// global_load_async_to_lds_b128 (ASYNCcnt) shared by all 8 waves of a block.
// ---------------------------------------------------------------------------

typedef __bf16 bf16_t;
typedef __attribute__((ext_vector_type(16))) __bf16 v16bf;
typedef __attribute__((ext_vector_type(8)))  __bf16 v8bf;
typedef __attribute__((ext_vector_type(8)))  float  v8f;
typedef __attribute__((ext_vector_type(4)))  float  v4f;

#define N_NODES 20000
#define N_PAD   20096        // 157 * 128
#define N_EDGES 160000
#define D_REAL  1000
#define D_PAD   1024
#define NLAYERS 3

// ---- CDNA5 async copy: global -> LDS, 16B per lane, tracked by ASYNCcnt ----
__device__ __forceinline__ void async_copy_b128(unsigned lds_addr, const void* gaddr) {
    asm volatile("global_load_async_to_lds_b128 %0, %1, off"
                 :: "v"(lds_addr), "v"(gaddr)
                 : "memory");
}
__device__ __forceinline__ void wait_async0() {
    asm volatile("s_wait_asynccnt 0x0" ::: "memory");
}

// ---- WMMA fragment helpers (layouts per CDNA5 ISA 7.12.2, wave32) ----------

// A (16x32 bf16, row-major source, ld elements): lane<16 -> row=lane,
// K = kb+{0..7, 16..23}; lane>=16 -> row=lane-16, K = kb+{8..15, 24..31}.
__device__ __forceinline__ v16bf load_a_bf16(const bf16_t* __restrict__ base,
                                             int row, int ld, int kb, int lane) {
    const int hsel = lane >> 4;
    const bf16_t* p = base + (size_t)row * ld + kb + hsel * 8;
    v8bf lo = *(const v8bf*)(p);
    v8bf hi = *(const v8bf*)(p + 16);
    v16bf a;
#pragma unroll
    for (int i = 0; i < 8; ++i) { a[i] = lo[i]; a[i + 8] = hi[i]; }
    return a;
}

// Same A fragment, converting from an f32 source on the fly.
__device__ __forceinline__ v16bf load_a_f32cvt(const float* __restrict__ base,
                                               int row, int ld, int kb, int lane) {
    const int hsel = lane >> 4;
    const float* p = base + (size_t)row * ld + kb + hsel * 8;
    v4f f0 = *(const v4f*)(p);
    v4f f1 = *(const v4f*)(p + 4);
    v4f f2 = *(const v4f*)(p + 16);
    v4f f3 = *(const v4f*)(p + 20);
    v16bf a;
#pragma unroll
    for (int i = 0; i < 4; ++i) {
        a[i]      = (__bf16)f0[i];
        a[i + 4]  = (__bf16)f1[i];
        a[i + 8]  = (__bf16)f2[i];
        a[i + 12] = (__bf16)f3[i];
    }
    return a;
}

// B fragment from an LDS tile holding Bt[c_local][k_local] rows of 32 bf16:
// lane<16 -> col=c, K={0..7,16..23} ; lane>=16 -> col=c, K={8..15,24..31}.
__device__ __forceinline__ v16bf load_b_lds(const bf16_t* lds_tile, int c_local, int lane) {
    const int hsel = lane >> 4;
    const bf16_t* p = lds_tile + c_local * 32 + hsel * 8;
    v8bf lo = *(const v8bf*)(p);
    v8bf hi = *(const v8bf*)(p + 16);
    v16bf b;
#pragma unroll
    for (int i = 0; i < 8; ++i) { b[i] = lo[i]; b[i + 8] = hi[i]; }
    return b;
}

__device__ __forceinline__ v8f wmma_bf16(v16bf a, v16bf b, v8f c) {
    return __builtin_amdgcn_wmma_f32_16x16x32_bf16(
        false, a, false, b, (short)0, c, false, false);
}

__device__ __forceinline__ float sigmoidf_(float x) {
    return 1.f / (1.f + __expf(-x));
}

// ---- padding / conversion kernels ------------------------------------------

__global__ void __launch_bounds__(256)
pad_x_kernel(const float* __restrict__ x, float* __restrict__ hf,
             bf16_t* __restrict__ hb) {
    int i = blockIdx.x * 256 + threadIdx.x;          // over N_PAD*D_PAD
    int row = i >> 10, col = i & 1023;
    float v = (row < N_NODES && col < D_REAL) ? x[(size_t)row * D_REAL + col] : 0.f;
    hf[i] = v;
    hb[i] = (__bf16)v;
}

// weight[l][k][c] (row-major [D,D]) -> Wt[l][c][k] padded bf16 [1024][1024]
__global__ void __launch_bounds__(256)
pad_wt_kernel(const float* __restrict__ w, bf16_t* __restrict__ wt) {
    int i = blockIdx.x * 256 + threadIdx.x;          // over 3*1024*1024
    int l = i >> 20;
    int rem = i & ((1 << 20) - 1);
    int c = rem >> 10, k = rem & 1023;
    float v = (c < D_REAL && k < D_REAL)
                  ? w[(size_t)l * D_REAL * D_REAL + (size_t)k * D_REAL + c]
                  : 0.f;
    wt[i] = (__bf16)v;
}

// w_gate[(g*1000+d)][k] -> out[(g*1024+d)][k] padded bf16 (already B-friendly)
__global__ void __launch_bounds__(256)
pad_gate_kernel(const float* __restrict__ w, bf16_t* __restrict__ o) {
    int i = blockIdx.x * 256 + threadIdx.x;          // over 3*1024*1024
    int cpad = i >> 10, k = i & 1023;
    int g = cpad >> 10;
    int d = cpad & 1023;
    float v = (d < D_REAL && k < D_REAL)
                  ? w[((size_t)g * D_REAL + d) * D_REAL + k]
                  : 0.f;
    o[i] = (__bf16)v;
}

__global__ void __launch_bounds__(256)
zero_f32_kernel(float* __restrict__ p) {            // float4 per thread
    int i = blockIdx.x * 256 + threadIdx.x;
    ((v4f*)p)[i] = v4f{0.f, 0.f, 0.f, 0.f};
}

// ---- m = h @ W[l] ----------------------------------------------------------
// Block: 8 waves = 128 rows x 64 cols. B tile (64 cols x 32 k = 4KB bf16)
// staged in double-buffered LDS via async copy; all 8 waves share it.

__global__ void __launch_bounds__(256)
gemm_m_kernel(const bf16_t* __restrict__ H, const bf16_t* __restrict__ Wt,
              bf16_t* __restrict__ M) {
    __shared__ __align__(64) char smemB[2 * 4096];
    const int tid = threadIdx.x;
    const int lane = tid & 31;
    const int wave = tid >> 5;
    const int rowBase = (blockIdx.x * 8 + wave) * 16;   // 157*8 = 1256 tiles
    const int colBase = blockIdx.y * 64;                // 16 col groups
    const int row = rowBase + (lane & 15);
    const unsigned lds0 = (unsigned)(size_t)(void*)smemB;

    // per-thread async chunk: c_local = tid>>2 (0..63), seg = tid&3
    const int cl = tid >> 2, seg = tid & 3;
    const char* gsrc = (const char*)(Wt + (size_t)(colBase + cl) * D_PAD) + seg * 16;

    v8f acc[4];
#pragma unroll
    for (int t = 0; t < 4; ++t) acc[t] = v8f{0.f,0.f,0.f,0.f,0.f,0.f,0.f,0.f};

    // prologue: stage first B chunk, prefetch first A fragment
    async_copy_b128(lds0 + cl * 64 + seg * 16, gsrc);
    v16bf a_cur = load_a_bf16(H, row, D_PAD, 0, lane);

    for (int kb = 0; kb < D_PAD; kb += 32) {
        const int sel = (kb >> 5) & 1;
        wait_async0();
        __syncthreads();                               // B[sel] ready for all
        v16bf a_next = a_cur;
        if (kb + 32 < D_PAD) {
            async_copy_b128(lds0 + (sel ^ 1) * 4096 + cl * 64 + seg * 16,
                            gsrc + (size_t)(kb + 32) * 2);
            a_next = load_a_bf16(H, row, D_PAD, kb + 32, lane);
        }
        const bf16_t* btile = (const bf16_t*)(smemB + sel * 4096);
#pragma unroll
        for (int t = 0; t < 4; ++t) {
            v16bf b = load_b_lds(btile, t * 16 + (lane & 15), lane);
            acc[t] = wmma_bf16(a_cur, b, acc[t]);
        }
        a_cur = a_next;
    }

    const int hsel = lane >> 4;
#pragma unroll
    for (int t = 0; t < 4; ++t) {
        int cc = colBase + t * 16 + (lane & 15);
#pragma unroll
        for (int j = 0; j < 8; ++j) {
            int r = rowBase + j + hsel * 8;
            M[(size_t)r * D_PAD + cc] =
                (cc < D_REAL) ? (__bf16)acc[t][j] : (__bf16)0.f;
        }
    }
}

// ---- edge scatter: agg[dst] += m[src] * ew  (f32 atomics, L2-resident) -----

__global__ void __launch_bounds__(256)
scatter_kernel(const bf16_t* __restrict__ M, const int* __restrict__ src,
               const int* __restrict__ dst, const float* __restrict__ ew,
               float* __restrict__ agg) {
    const int e = blockIdx.x;
    const int t = threadIdx.x;
    const int s = src[e];
    const int d = dst[e];
    const float w = ew[e];
    const bf16_t* mrow = M + (size_t)s * D_PAD;
    float* arow = agg + (size_t)d * D_PAD;
#pragma unroll
    for (int i = 0; i < 4; ++i) {
        int f = t + i * 256;
        if (f < D_REAL) atomicAdd(&arow[f], (float)mrow[f] * w);
    }
}

// ---- fused GRU cell --------------------------------------------------------
// Block: 8 waves = 128 rows x 16 feats. Six 16x32 gate B tiles (6KB bf16)
// staged in double-buffered LDS; 6 WMMA chains per wave; gates applied on
// the accumulator layout directly.

__global__ void __launch_bounds__(256)
gru_kernel(const float* __restrict__ AGG, const bf16_t* __restrict__ Hb,
           const float* __restrict__ Hf,
           const bf16_t* __restrict__ Wih, const bf16_t* __restrict__ Whh,
           const float* __restrict__ bih, const float* __restrict__ bhh,
           float* __restrict__ Hf_out, bf16_t* __restrict__ Hb_out) {
    __shared__ __align__(64) char smemB[2 * 6144];
    const int tid = threadIdx.x;
    const int lane = tid & 31;
    const int wave = tid >> 5;
    const int rowBase = (blockIdx.x * 8 + wave) * 16;
    const int dBase = blockIdx.y * 16;                 // 64 feat tiles
    const int row = rowBase + (lane & 15);
    const int dcol = dBase + (lane & 15);
    const unsigned lds0 = (unsigned)(size_t)(void*)smemB;

    // async chunks: 6 tiles * 64 chunks(16B) = 384; threads cover 256+128
    const char* gsrc[2];
    unsigned loff[2];
    int nchunk = 0;
#pragma unroll
    for (int c = 0; c < 2; ++c) {
        int i = tid + c * 256;
        if (i < 384) {
            int t = i >> 6;                 // 0..5 (0-2: Wih r,z,n; 3-5: Whh)
            int j = i & 63;
            int cl = j >> 2, seg = j & 3;
            const bf16_t* mat = (t < 3) ? Wih : Whh;
            int g = (t < 3) ? t : (t - 3);
            gsrc[nchunk] = (const char*)(mat + ((size_t)g * D_PAD + dBase + cl) * D_PAD)
                           + seg * 16;
            loff[nchunk] = t * 1024 + cl * 64 + seg * 16;
            ++nchunk;
        }
    }

    v8f aR{}, aZ{}, aN{}, hR{}, hZ{}, hN{};

    for (int c = 0; c < nchunk; ++c) async_copy_b128(lds0 + loff[c], gsrc[c]);
    v16bf am_cur = load_a_f32cvt(AGG, row, D_PAD, 0, lane);
    v16bf ah_cur = load_a_bf16(Hb, row, D_PAD, 0, lane);

    for (int kb = 0; kb < D_PAD; kb += 32) {
        const int sel = (kb >> 5) & 1;
        wait_async0();
        __syncthreads();
        v16bf am_next = am_cur, ah_next = ah_cur;
        if (kb + 32 < D_PAD) {
            for (int c = 0; c < nchunk; ++c)
                async_copy_b128(lds0 + (sel ^ 1) * 6144 + loff[c],
                                gsrc[c] + (size_t)(kb + 32) * 2);
            am_next = load_a_f32cvt(AGG, row, D_PAD, kb + 32, lane);
            ah_next = load_a_bf16(Hb, row, D_PAD, kb + 32, lane);
        }
        const bf16_t* btile = (const bf16_t*)(smemB + sel * 6144);
        const int cl = lane & 15;
        aR = wmma_bf16(am_cur, load_b_lds(btile + 0 * 512, cl, lane), aR);
        aZ = wmma_bf16(am_cur, load_b_lds(btile + 1 * 512, cl, lane), aZ);
        aN = wmma_bf16(am_cur, load_b_lds(btile + 2 * 512, cl, lane), aN);
        hR = wmma_bf16(ah_cur, load_b_lds(btile + 3 * 512, cl, lane), hR);
        hZ = wmma_bf16(ah_cur, load_b_lds(btile + 4 * 512, cl, lane), hZ);
        hN = wmma_bf16(ah_cur, load_b_lds(btile + 5 * 512, cl, lane), hN);
        am_cur = am_next;
        ah_cur = ah_next;
    }

    const int hsel = lane >> 4;
    const bool inb = (dcol < D_REAL);
    const float bir = inb ? bih[0 * D_REAL + dcol] : 0.f;
    const float biz = inb ? bih[1 * D_REAL + dcol] : 0.f;
    const float bin = inb ? bih[2 * D_REAL + dcol] : 0.f;
    const float bhr = inb ? bhh[0 * D_REAL + dcol] : 0.f;
    const float bhz = inb ? bhh[1 * D_REAL + dcol] : 0.f;
    const float bhn = inb ? bhh[2 * D_REAL + dcol] : 0.f;

#pragma unroll
    for (int j = 0; j < 8; ++j) {
        const int r = rowBase + j + hsel * 8;
        const size_t idx = (size_t)r * D_PAD + dcol;
        const float hold = Hf[idx];
        const float rg = sigmoidf_((aR[j] + bir) + (hR[j] + bhr));
        const float zg = sigmoidf_((aZ[j] + biz) + (hZ[j] + bhz));
        const float ng = tanhf((aN[j] + bin) + rg * (hN[j] + bhn));
        float hn = (1.f - zg) * ng + zg * hold;
        if (!inb) hn = 0.f;                            // keep K-pad zero
        Hf_out[idx] = hn;
        Hb_out[idx] = (__bf16)hn;
    }
}

// ---- out[n] = sum_d relu(h[n,d]) * fc_w[d] + fc_b --------------------------

__global__ void __launch_bounds__(256)
final_kernel(const float* __restrict__ H, const float* __restrict__ fcw,
             const float* __restrict__ fcb, float* __restrict__ out) {
    const int lane = threadIdx.x & 31;
    const int wave = threadIdx.x >> 5;
    const int n = blockIdx.x * 8 + wave;
    float s = 0.f;
    const float* hrow = H + (size_t)n * D_PAD;
    for (int d = lane; d < D_REAL; d += 32) {
        float v = hrow[d];
        v = v > 0.f ? v : 0.f;
        s += v * fcw[d];
    }
#pragma unroll
    for (int off = 16; off > 0; off >>= 1) s += __shfl_xor(s, off, 32);
    if (lane == 0) out[n] = s + fcb[0];
}

// ---------------------------------------------------------------------------

extern "C" void kernel_launch(void* const* d_in, const int* in_sizes, int n_in,
                              void* d_out, int out_size, void* d_ws, size_t ws_size,
                              hipStream_t stream) {
    const float* x    = (const float*)d_in[0];
    const int*   ei   = (const int*)d_in[1];        // [2, E]
    const float* ew   = (const float*)d_in[2];
    const float* wgt  = (const float*)d_in[3];      // [3, D, D]
    const float* wih  = (const float*)d_in[4];      // [3D, D]
    const float* whh  = (const float*)d_in[5];      // [3D, D]
    const float* bih  = (const float*)d_in[6];
    const float* bhh  = (const float*)d_in[7];
    const float* fcw  = (const float*)d_in[8];
    const float* fcb  = (const float*)d_in[9];
    float* out = (float*)d_out;

    const int* src = ei;
    const int* dst = ei + N_EDGES;

    // workspace carve-up (all sizes multiples of 256B)
    char* p = (char*)d_ws;
    float*  h0f  = (float*)p;  p += (size_t)N_PAD * D_PAD * 4;
    float*  h1f  = (float*)p;  p += (size_t)N_PAD * D_PAD * 4;
    bf16_t* h0b  = (bf16_t*)p; p += (size_t)N_PAD * D_PAD * 2;
    bf16_t* h1b  = (bf16_t*)p; p += (size_t)N_PAD * D_PAD * 2;
    bf16_t* mbf  = (bf16_t*)p; p += (size_t)N_PAD * D_PAD * 2;
    float*  agg  = (float*)p;  p += (size_t)N_PAD * D_PAD * 4;
    bf16_t* wtb  = (bf16_t*)p; p += (size_t)NLAYERS * D_PAD * D_PAD * 2;
    bf16_t* wihb = (bf16_t*)p; p += (size_t)3 * D_PAD * D_PAD * 2;
    bf16_t* whhb = (bf16_t*)p; p += (size_t)3 * D_PAD * D_PAD * 2;

    // per-call padding / precision conversion
    pad_x_kernel<<<(N_PAD * D_PAD) / 256, 256, 0, stream>>>(x, h0f, h0b);
    pad_wt_kernel<<<(NLAYERS * D_PAD * D_PAD) / 256, 256, 0, stream>>>(wgt, wtb);
    pad_gate_kernel<<<(3 * D_PAD * D_PAD) / 256, 256, 0, stream>>>(wih, wihb);
    pad_gate_kernel<<<(3 * D_PAD * D_PAD) / 256, 256, 0, stream>>>(whh, whhb);

    float*  hf[2] = {h0f, h1f};
    bf16_t* hb[2] = {h0b, h1b};
    int cur = 0;

    for (int l = 0; l < NLAYERS; ++l) {
        gemm_m_kernel<<<dim3(N_PAD / 128, 16), 256, 0, stream>>>(
            hb[cur], wtb + (size_t)l * D_PAD * D_PAD, mbf);
        zero_f32_kernel<<<(N_PAD * D_PAD / 4) / 256, 256, 0, stream>>>(agg);
        scatter_kernel<<<N_EDGES, 256, 0, stream>>>(mbf, src, dst, ew, agg);
        gru_kernel<<<dim3(N_PAD / 128, 64), 256, 0, stream>>>(
            agg, hb[cur], hf[cur], wihb, whhb, bih, bhh, hf[1 - cur], hb[1 - cur]);
        cur ^= 1;
    }

    final_kernel<<<2500, 256, 0, stream>>>(hf[cur], fcw, fcb, out);
}
